// BaseLoftqLinear_17068200034764
// MI455X (gfx1250) — matrix operand order repaired and checked by reading, not compiled, not run on hardware
//
#include <hip/hip_runtime.h>

// ---------------------------------------------------------------------------
// LoftQ linear (4-bit block-dequant + LoRA) for MI455X / gfx1250.
//
//   1) k_dequant_lora : dequant 4-bit weights, fold LORA_SCALE*(B@A) into them,
//                       emit W_eff as f16 row-major [OUT_F][IN_F].
//   2) k_pack_x       : convert x to f16 pre-swizzled into the CDNA5 WMMA
//                       A-fragment tile layout (16Mx32K tiles, 1KB each).
//   3) k_gemm         : 128x256 workgroup tile, 8 waves, 64x64 per wave
//                       (4x4 fragments), v_wmma_f32_16x16x32_f16 inner loop.
//                       Panels staged by the Tensor Data Mover
//                       (tensor_load_to_lds, TENSORcnt) with LDS double
//                       buffering; manual-copy fallback if builtin missing.
// ---------------------------------------------------------------------------

typedef _Float16 h16;
typedef __attribute__((ext_vector_type(16))) _Float16 v16h;
typedef __attribute__((ext_vector_type(8)))  _Float16 v8h;
typedef __attribute__((ext_vector_type(8)))  float    v8f;
typedef __attribute__((ext_vector_type(4)))  float    v4f;
typedef __attribute__((ext_vector_type(4)))  int      v4i;
typedef __attribute__((ext_vector_type(8)))  int      v8i;
typedef __attribute__((ext_vector_type(4)))  unsigned int v4u;

#define OUT_F   4096
#define IN_F    4096
#define M_TOK   2048          // 2*1024 tokens
#define LORA_R  16
#define LORA_SC (32.0f / 16.0f)

#define BM 128                // workgroup M tile
#define BN 256                // workgroup N tile
#define BK 32                 // K step (one f16 WMMA K)
#define KT (IN_F / BK)        // 128 K-tiles
#define B_ROW_H 40            // LDS B row stride in halves (32 + 8 pad = 80B)

#if defined(__has_builtin)
#if __has_builtin(__builtin_amdgcn_tensor_load_to_lds) && \
    __has_builtin(__builtin_amdgcn_s_wait_tensorcnt)
#define USE_TDM 1
#endif
#endif
#ifndef USE_TDM
#define USE_TDM 0
#endif

// ---------------------------------------------------------------------------
// Kernel 1: dequant + LoRA fold -> Wh[o][i] f16 row-major.
// One thread per 4 packed bytes = 8 weight elements (all in one scale block).
// ---------------------------------------------------------------------------
__global__ __launch_bounds__(256) void k_dequant_lora(
    const int*   __restrict__ qw,      // one byte (2 nibbles) per int
    const float* __restrict__ scales,  // [OUT_F*IN_F/64]
    const float* __restrict__ lA,      // [R][IN_F]
    const float* __restrict__ lB,      // [OUT_F][R]
    h16*         __restrict__ Wh)      // [OUT_F][IN_F]
{
    const int j  = blockIdx.x * 256 + threadIdx.x;  // 8-element group id
    const int f0 = j << 3;                          // first flat weight index
    const int o  = f0 >> 12;                        // / IN_F
    const int i0 = f0 & (IN_F - 1);

    const v4i q = *(const v4i*)(qw + (j << 2));
    const float scale = scales[f0 >> 6];

    float lb[LORA_R];
#pragma unroll
    for (int r = 0; r < LORA_R; ++r) lb[r] = lB[o * LORA_R + r];

    float lacc[8] = {0.f, 0.f, 0.f, 0.f, 0.f, 0.f, 0.f, 0.f};
#pragma unroll
    for (int r = 0; r < LORA_R; ++r) {
        const float* arow = lA + r * IN_F + i0;
        const v4f a0 = *(const v4f*)arow;
        const v4f a1 = *(const v4f*)(arow + 4);
        lacc[0] += lb[r] * a0.x;  lacc[1] += lb[r] * a0.y;
        lacc[2] += lb[r] * a0.z;  lacc[3] += lb[r] * a0.w;
        lacc[4] += lb[r] * a1.x;  lacc[5] += lb[r] * a1.y;
        lacc[6] += lb[r] * a1.z;  lacc[7] += lb[r] * a1.w;
    }

    const float step = 2.0f / 15.0f;   // table = linspace(-1,1,16)
    v8h w;
#pragma unroll
    for (int t = 0; t < 4; ++t) {
        const int b = ((const int*)&q)[t];
        const float vlo = -1.0f + (float)(b & 15) * step;
        const float vhi = -1.0f + (float)((b >> 4) & 15) * step;
        w[2 * t]     = (h16)(vlo * scale + LORA_SC * lacc[2 * t]);
        w[2 * t + 1] = (h16)(vhi * scale + LORA_SC * lacc[2 * t + 1]);
    }
    *(v8h*)(Wh + (size_t)o * IN_F + i0) = w;
}

// ---------------------------------------------------------------------------
// Kernel 2: x fp32 -> f16, pre-swizzled into WMMA A-fragment tile layout.
// Tile (mt,kt) = 16Mx32K = 1024B at (mt*KT+kt)*1024.  Within tile: group g,
// lane l: 16B chunk at g*512+l*16 = x[mt*16+(l&15)][kt*32+((l>=16)?8:0)+g*16..+8)
// ---------------------------------------------------------------------------
__global__ __launch_bounds__(256) void k_pack_x(
    const float* __restrict__ x, h16* __restrict__ Ah)
{
    const int c    = blockIdx.x * 256 + threadIdx.x;  // 16B chunk id
    const int tile = c >> 6;
    const int r    = c & 63;
    const int g    = r >> 5;
    const int l    = r & 31;
    const int mt   = tile >> 7;                       // / KT
    const int kt   = tile & (KT - 1);
    const int m    = mt * 16 + (l & 15);
    const int k    = kt * 32 + ((l >> 4) << 3) + g * 16;

    const float* src = x + (size_t)m * IN_F + k;
    const v4f x0 = *(const v4f*)src;
    const v4f x1 = *(const v4f*)(src + 4);
    v8h o;
    o[0] = (h16)x0.x; o[1] = (h16)x0.y; o[2] = (h16)x0.z; o[3] = (h16)x0.w;
    o[4] = (h16)x1.x; o[5] = (h16)x1.y; o[6] = (h16)x1.z; o[7] = (h16)x1.w;
    *(v8h*)(Ah + (size_t)c * 8) = o;
}

// ---------------------------------------------------------------------------
// TDM helper: one 2D tensor_load_to_lds per panel (6-arg builtin form).
// D# group0: count=1 | lds_addr | global_addr(57b) | type=2.
// D# group1: data_size=2B (+pad ctrl) | dims | tile dims | row stride.
// Groups 2/3 + trailing group zero (<=2D tensor).
// ---------------------------------------------------------------------------
#if USE_TDM
__device__ __forceinline__ void tdm_load_2d(
    unsigned lds_addr, unsigned long long gaddr,
    unsigned tensor_d0, unsigned tensor_d1,
    unsigned tile_d0, unsigned tile_d1,
    unsigned long long stride0_elems, unsigned pad_ctrl)
{
    v4u g0;
    g0.x = 1u;                                        // count=1 (valid, user)
    g0.y = lds_addr;                                  // LDS byte address
    g0.z = (unsigned)(gaddr & 0xFFFFFFFFu);           // global_addr[31:0]
    g0.w = (unsigned)((gaddr >> 32) & 0x1FFFFFFu)     // global_addr[56:32]
         | 0x80000000u;                               // type=2 ("image")
    v8i g1;
    g1[0] = (int)((1u << 16) | pad_ctrl);             // data_size=2B + pad
    g1[1] = (int)((tensor_d0 & 0xFFFFu) << 16);       // tensor_dim0[15:0]
    g1[2] = (int)((tensor_d0 >> 16) | ((tensor_d1 & 0xFFFFu) << 16));
    g1[3] = (int)((tensor_d1 >> 16) | (tile_d0 << 16));   // tile_dim0
    g1[4] = (int)(tile_d1 & 0xFFFFu);                 // tile_dim1, tile_dim2=0
    g1[5] = (int)(stride0_elems & 0xFFFFFFFFu);       // dim0 stride
    g1[6] = (int)((stride0_elems >> 32) & 0xFFFFu);   // stride hi, stride1=0
    g1[7] = 0;
    const v4i z4 = {0, 0, 0, 0};
    const v8i z8 = {0, 0, 0, 0, 0, 0, 0, 0};
    __builtin_amdgcn_tensor_load_to_lds(g0, g1, z4, z4, z8, 0);
}
#endif

// ---------------------------------------------------------------------------
// Kernel 3: GEMM  out[m][n] = sum_k Ah[m][k] * Wh[n][k] + bias[n]
// 256 threads = 8 waves (wave32). Wave tile 64x64 = 4x4 fragments.
// ---------------------------------------------------------------------------
__global__ __launch_bounds__(256, 1) void k_gemm(
    const h16*   __restrict__ Ah,
    const h16*   __restrict__ Wh,
    const float* __restrict__ bias,
    float*       __restrict__ out)
{
    __shared__ __align__(32) h16 ldsA[2][BM * BK];        // 2 x 8KB
    __shared__ __align__(32) h16 ldsB[2][BN * B_ROW_H];   // 2 x 20KB

    const int tid  = threadIdx.x;
    const int lane = tid & 31;
    const int w    = tid >> 5;
    const int wm   = w >> 2;                 // 0..1
    const int wn   = w & 3;                  // 0..3
    const int mt0  = blockIdx.y * (BM / 16); // first 16-row m-tile of this WG
    const int n0   = blockIdx.x * BN;

    v8f acc[4][4];
#pragma unroll
    for (int i = 0; i < 4; ++i)
#pragma unroll
        for (int jn = 0; jn < 4; ++jn) acc[i][jn] = (v8f)(0.0f);

#if USE_TDM
    // Prologue: wave 0 kicks off the DMA for kt=0 into buffer 0.
    if (tid < 32) {
        tdm_load_2d((unsigned)(unsigned long long)(uintptr_t)&ldsA[0][0],
                    (unsigned long long)(uintptr_t)(Ah + (size_t)mt0 * KT * 512),
                    512u, 128u, 512u, 8u, (unsigned long long)KT * 512u, 0u);
        tdm_load_2d((unsigned)(unsigned long long)(uintptr_t)&ldsB[0][0],
                    (unsigned long long)(uintptr_t)(Wh + (size_t)n0 * IN_F),
                    (unsigned)IN_F, (unsigned)OUT_F, (unsigned)BK, (unsigned)BN,
                    (unsigned long long)IN_F,
                    (1u << 20) | (3u << 22) | (3u << 25)); // pad 16B per 64B row
    }
#endif

    for (int kt = 0; kt < KT; ++kt) {
        const int buf = kt & 1;

#if USE_TDM
        __syncthreads();   // everyone done reading buffer 1-buf (iteration kt-1)
        if (tid < 32) {
            const int kn = (kt + 1 < KT) ? (kt + 1) : kt;  // keep wait pattern uniform
            tdm_load_2d((unsigned)(unsigned long long)(uintptr_t)&ldsA[1 - buf][0],
                        (unsigned long long)(uintptr_t)
                            (Ah + ((size_t)mt0 * KT + kn) * 512),
                        512u, 128u, 512u, 8u, (unsigned long long)KT * 512u, 0u);
            tdm_load_2d((unsigned)(unsigned long long)(uintptr_t)&ldsB[1 - buf][0],
                        (unsigned long long)(uintptr_t)
                            (Wh + (size_t)n0 * IN_F + (size_t)kn * BK),
                        (unsigned)IN_F, (unsigned)OUT_F, (unsigned)BK, (unsigned)BN,
                        (unsigned long long)IN_F,
                        (1u << 20) | (3u << 22) | (3u << 25));
            // 2 new ops outstanding; in-order TENSORcnt<=2 => kt's panels done.
            __builtin_amdgcn_s_wait_tensorcnt(2);
        }
        __syncthreads();   // kt's panels visible to all waves
#else
        __syncthreads();
#pragma unroll
        for (int q = 0; q < 2; ++q) {           // A: 512 x 16B chunks
            const int c = q * 256 + tid;
            const int t = c >> 6;
            const int r = c & 63;
            const v8h* src =
                (const v8h*)(Ah + ((size_t)(mt0 + t) * KT + kt) * 512) + r;
            ((v8h*)ldsA[buf])[c] = *src;
            __builtin_prefetch((const char*)src + 1024, 0, 1);
        }
#pragma unroll
        for (int q = 0; q < 4; ++q) {           // B: 1024 x 16B chunks
            const int c  = q * 256 + tid;
            const int n  = c >> 2;
            const int cc = c & 3;
            const v8h* src =
                (const v8h*)(Wh + (size_t)(n0 + n) * IN_F + kt * 32) + cc;
            *(v8h*)(ldsB[buf] + n * B_ROW_H + cc * 8) = *src;
            __builtin_prefetch((const char*)src + 64, 0, 1);
        }
        __syncthreads();
#endif

        // Fragment loads (per-lane contiguous thanks to pre-swizzle).
        const h16* lA = ldsA[buf];
        const h16* lB = ldsB[buf];
        v16h afr[4], bfr[4];
#pragma unroll
        for (int i = 0; i < 4; ++i) {
            const int t = wm * 4 + i;
            const v8h alo = *(const v8h*)(lA + t * 512 + lane * 8);
            const v8h ahi = *(const v8h*)(lA + t * 512 + 256 + lane * 8);
            afr[i] = __builtin_shufflevector(alo, ahi,
                0, 1, 2, 3, 4, 5, 6, 7, 8, 9, 10, 11, 12, 13, 14, 15);
        }
#pragma unroll
        for (int i = 0; i < 4; ++i) {
            const int nl = wn * 64 + i * 16 + (lane & 15);
            const h16* p = lB + nl * B_ROW_H + ((lane >> 4) << 3);
            const v8h blo = *(const v8h*)p;
            const v8h bhi = *(const v8h*)(p + 16);
            bfr[i] = __builtin_shufflevector(blo, bhi,
                0, 1, 2, 3, 4, 5, 6, 7, 8, 9, 10, 11, 12, 13, 14, 15);
        }

        // 16 WMMAs per K-step.
#pragma unroll
        for (int i = 0; i < 4; ++i)
#pragma unroll
            for (int jn = 0; jn < 4; ++jn)
                acc[i][jn] = __builtin_amdgcn_wmma_f32_16x16x32_f16(
                    false, afr[i], false, bfr[jn],
                    (short)0, acc[i][jn], false, false);
    }

    // Epilogue: C layout = VGPR r: lanes 0-15 -> M=r, lanes 16-31 -> M=r+8.
    const int m_w     = mt0 * 16 + wm * 64;
    const int n_w     = n0 + wn * 64;
    const int row_off = (lane >> 4) << 3;
    const int col     = lane & 15;
#pragma unroll
    for (int i = 0; i < 4; ++i) {
#pragma unroll
        for (int jn = 0; jn < 4; ++jn) {
            const int n  = n_w + jn * 16 + col;
            const float bv = bias[n];
#pragma unroll
            for (int r2 = 0; r2 < 8; ++r2) {
                const int m = m_w + i * 16 + r2 + row_off;
                out[(size_t)m * OUT_F + n] = acc[i][jn][r2] + bv;
            }
        }
    }
}

// ---------------------------------------------------------------------------
extern "C" void kernel_launch(void* const* d_in, const int* in_sizes, int n_in,
                              void* d_out, int out_size, void* d_ws, size_t ws_size,
                              hipStream_t stream) {
    const float* x      = (const float*)d_in[0];
    const int*   qw     = (const int*)d_in[1];
    const float* scales = (const float*)d_in[2];
    const float* bias   = (const float*)d_in[3];
    const float* lA     = (const float*)d_in[4];
    const float* lB     = (const float*)d_in[5];
    float*       out    = (float*)d_out;

    // Workspace: Ah (16MB swizzled f16 x) | Wh (32MB f16 W_eff)
    h16* Ah = (h16*)d_ws;
    h16* Wh = (h16*)((char*)d_ws + (size_t)M_TOK * IN_F * sizeof(h16));

    (void)in_sizes; (void)n_in; (void)out_size; (void)ws_size;

    k_dequant_lora<<<(OUT_F * IN_F / 8) / 256, 256, 0, stream>>>(qw, scales, lA, lB, Wh);
    k_pack_x<<<(M_TOK * IN_F / 8) / 256, 256, 0, stream>>>(x, Ah);

    dim3 grid(OUT_F / BN, M_TOK / BM);   // 16 x 16 = 256 workgroups
    k_gemm<<<grid, 256, 0, stream>>>(Ah, Wh, bias, out);
}
